// MultiHeadAttention_39333310497121
// MI455X (gfx1250) — matrix operand order repaired and checked by reading, not compiled, not run on hardware
//
#include <hip/hip_runtime.h>
#include <hip/hip_bf16.h>
#include <cstdint>
#include <cstddef>

typedef __attribute__((ext_vector_type(16))) __bf16 v16bf;
typedef __attribute__((ext_vector_type(8)))  float  v8f;

union BFrag { v16bf v; uint4 u[2]; };
union U8bf  { uint4 u; __bf16 b[8]; };

static __device__ __forceinline__ v8f wmma_bf16(v16bf a, v16bf b, v8f c) {
  // 8 args: (neg_a, A, neg_b, B, c_mod, C, reuse_a, reuse_b)
  return __builtin_amdgcn_wmma_f32_16x16x32_bf16(false, a, false, b, (short)0, c,
                                                 false, false);
}

// Async global -> LDS copy, 16 bytes per lane, tracked by ASYNCcnt.
static __device__ __forceinline__ void async_copy_b128(void* lds, const void* gsrc) {
  unsigned l = (unsigned)(uintptr_t)lds;               // flat low 32 = LDS offset
  unsigned long long g = (unsigned long long)(uintptr_t)gsrc;
  asm volatile("global_load_async_to_lds_b128 %0, %1, off"
               :: "v"(l), "v"(g) : "memory");
}
static __device__ __forceinline__ void wait_async0() {
  asm volatile("s_wait_asynccnt 0x0" ::: "memory");
}
static __device__ __forceinline__ void wait_ds0() {
  asm volatile("s_wait_dscnt 0x0" ::: "memory");
}

// ---------------------------------------------------------------------------
// GEMM: C[M,N] = A[M,K] @ B[K,N] * scale   (bf16 WMMA, f32 accumulate)
//  A_BF16: A is bf16 (streamed to LDS with async b128 copies); else fp32+cvt.
//  ROPE:   apply RoPE (head dim 64, position = row % L) in the epilogue.
//  OUT_BF16: store bf16, else fp32.
// Block tile 128x128, K-step 32, 256 threads = 8 waves in 4(M) x 2(N) grid.
// ---------------------------------------------------------------------------
template <int A_BF16, int ROPE, int OUT_BF16>
__global__ __launch_bounds__(256) void gemm_tile(
    const void* __restrict__ Ap, const float* __restrict__ Bw,
    void* __restrict__ Cp, int M, int N, int K, float scale, int L)
{
  __shared__ __bf16 sA[128 * 40];   // [row][k], stride 40 elems (80B, 16B-aligned)
  __shared__ __bf16 sB[128 * 40];   // [col][k] (transposed), stride 40

  const int tid  = threadIdx.x;
  const int lane = tid & 31;
  const int wave = tid >> 5;
  const int wm   = wave & 3;
  const int wn   = wave >> 2;
  const int hx   = (lane >> 4) & 1;
  const int lc   = lane & 15;
  const int m0   = blockIdx.y * 128;
  const int n0   = blockIdx.x * 128;

  v8f acc[2][4];
  const v8f vzero = {0.f,0.f,0.f,0.f,0.f,0.f,0.f,0.f};
  for (int i = 0; i < 2; ++i)
    for (int j = 0; j < 4; ++j)
      acc[i][j] = vzero;

  for (int kt = 0; kt < K; kt += 32) {
    if (A_BF16) {
      // A tile 128x32 bf16 = 512 x 16B chunks, 2 async copies per thread
      const __bf16* A = (const __bf16*)Ap;
#pragma unroll
      for (int i = 0; i < 2; ++i) {
        int idx = tid + i * 256;
        int r = idx >> 2, c8 = idx & 3;       // 4 chunks of 8 bf16 per row
        async_copy_b128(&sA[r * 40 + c8 * 8],
                        A + (size_t)(m0 + r) * K + kt + c8 * 8);
      }
    } else {
      // A tile 128x32 fp32 = 1024 float4, 4 per thread, convert to bf16
      const float* A = (const float*)Ap;
#pragma unroll
      for (int i = 0; i < 4; ++i) {
        int idx = tid + i * 256;
        int r = idx >> 3, c4 = idx & 7;
        float4 f = *(const float4*)(A + (size_t)(m0 + r) * K + kt + c4 * 4);
        __bf16* d = &sA[r * 40 + c4 * 4];
        d[0] = (__bf16)f.x; d[1] = (__bf16)f.y; d[2] = (__bf16)f.z; d[3] = (__bf16)f.w;
      }
    }
    // B tile: 32 rows(k) x 128 cols fp32, store transposed [col][k] as bf16
#pragma unroll
    for (int i = 0; i < 4; ++i) {
      int idx = tid + i * 256;
      int kr = idx >> 5, c4 = idx & 31;
      float4 f = *(const float4*)(Bw + (size_t)(kt + kr) * N + n0 + c4 * 4);
      sB[(c4 * 4 + 0) * 40 + kr] = (__bf16)f.x;
      sB[(c4 * 4 + 1) * 40 + kr] = (__bf16)f.y;
      sB[(c4 * 4 + 2) * 40 + kr] = (__bf16)f.z;
      sB[(c4 * 4 + 3) * 40 + kr] = (__bf16)f.w;
    }
    if (A_BF16) wait_async0();
    __syncthreads();

    BFrag a[2], b[4];
#pragma unroll
    for (int i = 0; i < 2; ++i) {
      int row = wm * 32 + i * 16 + lc;
      // A layout (16-bit 16x32): K chunks at half*8 and 16+half*8
      a[i].u[0] = *(const uint4*)&sA[row * 40 + hx * 8];
      a[i].u[1] = *(const uint4*)&sA[row * 40 + 16 + hx * 8];
    }
#pragma unroll
    for (int j = 0; j < 4; ++j) {
      int col = wn * 64 + j * 16 + lc;
      // B layout (16-bit 32x16): K chunks at half*8 and 16+half*8
      b[j].u[0] = *(const uint4*)&sB[col * 40 + hx * 8];
      b[j].u[1] = *(const uint4*)&sB[col * 40 + 16 + hx * 8];
    }
#pragma unroll
    for (int i = 0; i < 2; ++i)
#pragma unroll
      for (int j = 0; j < 4; ++j)
        acc[i][j] = wmma_bf16(a[i].v, b[j].v, acc[i][j]);
    __syncthreads();
  }

  // Epilogue. C layout: VGPR r -> row hx*8+r, col = lane&15 of 16-wide subtile.
  // RoPE pairs (2p, 2p+1) are adjacent lanes -> partner via shfl_xor(1).
#pragma unroll
  for (int i = 0; i < 2; ++i)
#pragma unroll
    for (int j = 0; j < 4; ++j) {
      const int col = n0 + wn * 64 + j * 16 + lc;
      float inv = 0.f;
      if (ROPE) {
        int p = (col & 63) >> 1;   // head dim 64
        inv = expf(-9.2103403719761836f * ((2.0f * (float)p) / 64.0f));
      }
#pragma unroll
      for (int r = 0; r < 8; ++r) {
        const int row = m0 + wm * 32 + i * 16 + hx * 8 + r;
        float v = acc[i][j][r] * scale;
        if (ROPE) {
          float ang = (float)(row % L) * inv;
          float sn = sinf(ang), cs = cosf(ang);
          float pr = __shfl_xor(v, 1, 32);
          v = (lc & 1) ? (v * cs + pr * sn) : (v * cs - pr * sn);
        }
        if (OUT_BF16) ((__bf16*)Cp)[(size_t)row * N + col] = (__bf16)v;
        else          ((float*)Cp)[(size_t)row * N + col] = v;
      }
    }
}

// ---------------------------------------------------------------------------
// Causal flash attention. One block = 128 query rows of one (b, h).
// q/k/v/y are bf16 in (b, l, D) layout, head slice at col h*64. dh = 64.
// K tiles streamed global->LDS with async b128 copies (ASYNCcnt).
// ---------------------------------------------------------------------------
__global__ __launch_bounds__(256) void flash_attn(
    const __bf16* __restrict__ q, const __bf16* __restrict__ kbuf,
    const __bf16* __restrict__ vbuf, __bf16* __restrict__ y,
    int B, int L, int D, int H)
{
  const int dh = 64;
  __shared__ __bf16 sK[32 * 72];    // [key][dh]   stride 72 (144B, 16B-aligned)
  __shared__ __bf16 sV[64 * 40];    // [dh][key]   (transposed) stride 40
  __shared__ __bf16 sP[8][16 * 40]; // per-wave P [m][kp] stride 40

  const int tid  = threadIdx.x;
  const int lane = tid & 31;
  const int wave = tid >> 5;
  const int hx   = (lane >> 4) & 1;
  const int lc   = lane & 15;
  const int bh   = blockIdx.y;
  const int b    = bh / H;
  const int h    = bh % H;
  const int row0 = blockIdx.x * 128;
  const int qrow = row0 + wave * 16;

  const __bf16* qbase = q    + (size_t)b * L * D + (size_t)h * dh;
  const __bf16* kbase = kbuf + (size_t)b * L * D + (size_t)h * dh;
  const __bf16* vbase = vbuf + (size_t)b * L * D + (size_t)h * dh;

  // Q A-frags: lane holds row m = lc; two 16x32 frags cover dh=64
  BFrag qa[2];
  {
    const __bf16* qr = qbase + (size_t)(qrow + lc) * D;
#pragma unroll
    for (int kf = 0; kf < 2; ++kf) {
      qa[kf].u[0] = *(const uint4*)(qr + kf * 32 + hx * 8);
      qa[kf].u[1] = *(const uint4*)(qr + kf * 32 + 16 + hx * 8);
    }
  }

  const v8f vzero = {0.f,0.f,0.f,0.f,0.f,0.f,0.f,0.f};
  v8f o[4];
  float mrow[8], lrow[8];
#pragma unroll
  for (int j = 0; j < 4; ++j) o[j] = vzero;
#pragma unroll
  for (int r = 0; r < 8; ++r) { mrow[r] = -1e30f; lrow[r] = 0.0f; }

  // Per-thread tile-load mapping: 32 keys x 8 chunks(16B) = 256 = blockDim
  const int tkey = tid >> 3;
  const int tc8  = tid & 7;

  const int njt = blockIdx.x * 4 + 4;  // 32-key tiles up to the diagonal
  for (int jt = 0; jt < njt; ++jt) {
    const int jbase = jt * 32;
    __syncthreads();  // previous tile fully consumed before overwrite

    // K tile: async copy, layout preserved [key][dh]
    async_copy_b128(&sK[tkey * 72 + tc8 * 8],
                    kbase + (size_t)(jbase + tkey) * D + tc8 * 8);
    // V tile: load 8 bf16 and scatter transposed [dh][key]
    {
      U8bf t;
      t.u = *(const uint4*)(vbase + (size_t)(jbase + tkey) * D + tc8 * 8);
#pragma unroll
      for (int e = 0; e < 8; ++e)
        sV[(tc8 * 8 + e) * 40 + tkey] = t.b[e];
    }
    wait_async0();
    __syncthreads();

    // S = Q @ K^T : 16 x 32, two C tiles
    v8f sc[2];
#pragma unroll
    for (int n0 = 0; n0 < 2; ++n0) {
      int key = n0 * 16 + lc;
      BFrag kb;
      kb.u[0] = *(const uint4*)&sK[key * 72 + hx * 8];
      kb.u[1] = *(const uint4*)&sK[key * 72 + 16 + hx * 8];
      v8f cc = wmma_bf16(qa[0].v, kb.v, vzero);
      kb.u[0] = *(const uint4*)&sK[key * 72 + 32 + hx * 8];
      kb.u[1] = *(const uint4*)&sK[key * 72 + 32 + 16 + hx * 8];
      sc[n0] = wmma_bf16(qa[1].v, kb.v, cc);
    }

    // Online softmax (rows hx*8+r, cols jbase + n0*16 + lc)
    float pv[2][8], mx[8];
#pragma unroll
    for (int r = 0; r < 8; ++r) {
      int rowg = qrow + hx * 8 + r;
      float s0 = (jbase + lc      <= rowg) ? sc[0][r] : -1e30f;
      float s1 = (jbase + 16 + lc <= rowg) ? sc[1][r] : -1e30f;
      pv[0][r] = s0; pv[1][r] = s1;
      mx[r] = fmaxf(s0, s1);
    }
#pragma unroll
    for (int off = 1; off < 16; off <<= 1)
#pragma unroll
      for (int r = 0; r < 8; ++r)
        mx[r] = fmaxf(mx[r], __shfl_xor(mx[r], off, 32));

    float alpha[8], rs[8];
#pragma unroll
    for (int r = 0; r < 8; ++r) {
      float mn = fmaxf(mrow[r], mx[r]);
      alpha[r] = __expf(mrow[r] - mn);
      mrow[r] = mn;
    }
#pragma unroll
    for (int r = 0; r < 8; ++r) {
      pv[0][r] = __expf(pv[0][r] - mrow[r]);   // masked -> exp(~-1e30) = 0
      pv[1][r] = __expf(pv[1][r] - mrow[r]);
      rs[r] = pv[0][r] + pv[1][r];
    }
#pragma unroll
    for (int off = 1; off < 16; off <<= 1)
#pragma unroll
      for (int r = 0; r < 8; ++r)
        rs[r] += __shfl_xor(rs[r], off, 32);
#pragma unroll
    for (int r = 0; r < 8; ++r) lrow[r] = lrow[r] * alpha[r] + rs[r];
#pragma unroll
    for (int j = 0; j < 4; ++j)
#pragma unroll
      for (int r = 0; r < 8; ++r) o[j][r] *= alpha[r];

    // P (C layout) -> per-wave LDS -> A-frag layout
    __bf16* pw = &sP[wave][0];
#pragma unroll
    for (int n0 = 0; n0 < 2; ++n0)
#pragma unroll
      for (int r = 0; r < 8; ++r)
        pw[(hx * 8 + r) * 40 + n0 * 16 + lc] = (__bf16)pv[n0][r];
    wait_ds0();

    BFrag pf;
    pf.u[0] = *(const uint4*)&pw[lc * 40 + hx * 8];
    pf.u[1] = *(const uint4*)&pw[lc * 40 + 16 + hx * 8];

    // O += P @ V : four 16x16 C tiles across dh
#pragma unroll
    for (int j = 0; j < 4; ++j) {
      int dr = j * 16 + lc;
      BFrag vb;
      vb.u[0] = *(const uint4*)&sV[dr * 40 + hx * 8];
      vb.u[1] = *(const uint4*)&sV[dr * 40 + 16 + hx * 8];
      o[j] = wmma_bf16(pf.v, vb.v, o[j]);
    }
  }

  // Normalize and write y (bf16, (b,l,D) layout)
#pragma unroll
  for (int j = 0; j < 4; ++j)
#pragma unroll
    for (int r = 0; r < 8; ++r) {
      int rowg = qrow + hx * 8 + r;
      y[((size_t)b * L + rowg) * D + h * dh + j * 16 + lc] =
          (__bf16)(o[j][r] / lrow[r]);
    }
}

// ---------------------------------------------------------------------------
extern "C" void kernel_launch(void* const* d_in, const int* in_sizes, int n_in,
                              void* d_out, int out_size, void* d_ws, size_t ws_size,
                              hipStream_t stream) {
  const float* x  = (const float*)d_in[0];
  const float* Wq = (const float*)d_in[1];
  const float* Wk = (const float*)d_in[2];
  const float* Wv = (const float*)d_in[3];
  const float* Wo = (const float*)d_in[4];
  float* out = (float*)d_out;

  const int B = 2, L = 2048, D = 1024, H = 16;
  const int M = B * L;               // 4096
  const size_t n_elem = (size_t)M * D;

  __bf16* qb = (__bf16*)d_ws;
  __bf16* kb = qb + n_elem;
  __bf16* vb = kb + n_elem;
  __bf16* yb = vb + n_elem;

  dim3 gGemm(D / 128, M / 128);      // (8, 32)
  // q = rope(x@Wq * 1/sqrt(D)); k = rope(x@Wk); v = x@Wv   (bf16 outputs)
  gemm_tile<0, 1, 1><<<gGemm, 256, 0, stream>>>(x, Wq, qb, M, D, D, 0.03125f, L);
  gemm_tile<0, 1, 1><<<gGemm, 256, 0, stream>>>(x, Wk, kb, M, D, D, 1.0f, L);
  gemm_tile<0, 0, 1><<<gGemm, 256, 0, stream>>>(x, Wv, vb, M, D, D, 1.0f, L);

  // Causal flash attention -> yb (bf16)
  flash_attn<<<dim3(L / 128, B * H), 256, 0, stream>>>(qb, kb, vb, yb, B, L, D, H);

  // out = yb @ Wo (fp32 output, bf16 A streamed via async LDS copies)
  gemm_tile<1, 0, 0><<<gGemm, 256, 0, stream>>>(yb, Wo, out, M, D, D, 1.0f, L);
}
